// DeformableSubspaceModulatedConv2d_74543452390134
// MI455X (gfx1250) — matrix operand order repaired and checked by reading, not compiled, not run on hardware
//
#include <hip/hip_runtime.h>

// ---------------------------------------------------------------------------
// DeformableSubspaceModulatedConv2d for MI455X (gfx1250, wave32, WMMA bf16)
//
// B=16, CIN=512, COUT=512, K=3, H=W=64, STYLE=512, BASIS=8, DIRS=8
// ---------------------------------------------------------------------------

#define BB    16
#define CIN   512
#define COUT  512
#define HH    64
#define WW    64
#define KK    3
#define TAPS  9                     // 3x3
#define STYLE 512
#define BASIS 8
#define PTOT  (COUT*CIN*TAPS)       // 2,359,296 elements per basis vector
#define HP    66                    // padded spatial extent (halo of 1)

typedef __attribute__((ext_vector_type(16))) __bf16 v16bf;
typedef __attribute__((ext_vector_type(8)))  float  v8f;
typedef __attribute__((ext_vector_type(8)))  unsigned short us8;   // 16 bytes
typedef int i32x4 __attribute__((vector_size(16)));                // for async builtin

union FragA { v16bf bf; us8 u[2]; };

__device__ __forceinline__ unsigned short f32_to_bf16(float f) {
  unsigned int u = __float_as_uint(f);
  u = (u + 0x7FFFu + ((u >> 16) & 1u)) >> 16;   // round-to-nearest-even
  return (unsigned short)u;
}

// ---- CDNA5 async memory->LDS copy (ASYNCcnt path), with safe fallback ------
#if __has_builtin(__builtin_amdgcn_global_load_async_to_lds_b128)
#define HAVE_ASYNC_LDS 1
#else
#define HAVE_ASYNC_LDS 0
#endif

__device__ __forceinline__ void copy16_g2lds(const unsigned short* gsrc,
                                             unsigned short* ldst) {
#if HAVE_ASYNC_LDS
  __builtin_amdgcn_global_load_async_to_lds_b128(
      (__attribute__((address_space(1))) i32x4*)(unsigned short*)gsrc,
      (__attribute__((address_space(3))) i32x4*)ldst, 0, 0);
#else
  *(uint4*)ldst = *(const uint4*)gsrc;
#endif
}

#if HAVE_ASYNC_LDS
#if __has_builtin(__builtin_amdgcn_s_wait_asynccnt)
#define WAIT_ASYNC(n) __builtin_amdgcn_s_wait_asynccnt(n)
#else
#define WAIT_ASYNC(n) asm volatile("s_wait_asynccnt %0" :: "i"(n) : "memory")
#endif
#else
#define WAIT_ASYNC(n) ((void)0)
#endif

// ---------------------------------------------------------------------------
// K0: style modulation  s[b,i] = style[b,:] . mod_w[i,:] + mod_b[i]
// ---------------------------------------------------------------------------
__global__ void style_kernel(const float* __restrict__ style,
                             const float* __restrict__ mod_w,
                             const float* __restrict__ mod_b,
                             float* __restrict__ s) {
  int idx = blockIdx.x * 256 + threadIdx.x;      // 16*512 = 8192
  int b = idx >> 9, i = idx & 511;
  const float* st = style + b * STYLE;
  const float* mw = mod_w + (long long)i * STYLE;
  float acc = mod_b[i];
  for (int k = 0; k < STYLE; ++k) acc += st[k] * mw[k];
  s[idx] = acc;
}

// ---------------------------------------------------------------------------
// K1: normsq[b] = sum_p (sum_k coef[b,k]*bv[k,p])^2     (delta recomputed later)
// ---------------------------------------------------------------------------
__global__ void normsq_kernel(const float* __restrict__ bv,
                              const float* __restrict__ shifts_coords,
                              const int*   __restrict__ dirs,
                              float* __restrict__ normsq) {
  __shared__ float coef[BB][BASIS];
  __shared__ float red[256];
  int tid = threadIdx.x;
  if (tid < BB * BASIS) {
    int b = tid / BASIS, k = tid % BASIS;
    coef[b][k] = shifts_coords[dirs[b] * BASIS + k];
  }
  __syncthreads();

  float acc[BB];
#pragma unroll
  for (int b = 0; b < BB; ++b) acc[b] = 0.f;

  int stride = gridDim.x * blockDim.x;
  for (int p = blockIdx.x * blockDim.x + tid; p < PTOT; p += stride) {
    float bvv[BASIS];
#pragma unroll
    for (int k = 0; k < BASIS; ++k) bvv[k] = bv[(long long)k * PTOT + p];
#pragma unroll
    for (int b = 0; b < BB; ++b) {
      float d = 0.f;
#pragma unroll
      for (int k = 0; k < BASIS; ++k) d += coef[b][k] * bvv[k];
      acc[b] += d * d;
    }
  }

  for (int b = 0; b < BB; ++b) {
    __syncthreads();
    red[tid] = acc[b];
    __syncthreads();
    for (int off = 128; off > 0; off >>= 1) {
      if (tid < off) red[tid] += red[tid + off];
      __syncthreads();
    }
    if (tid == 0) atomicAdd(&normsq[b], red[0]);
  }
}

// ---------------------------------------------------------------------------
// K2: per-sample weight build (mod + demod) -> bf16, layout [b][o][tap][i]
// one block per (b,o)
// ---------------------------------------------------------------------------
__global__ void build_weights_kernel(const float* __restrict__ weight,
                                     const float* __restrict__ bv,
                                     const float* __restrict__ shifts_coords,
                                     const int*   __restrict__ dirs,
                                     const float* __restrict__ shifts,
                                     const float* __restrict__ s,
                                     const float* __restrict__ normsq,
                                     unsigned short* __restrict__ wgt) {
  const int o = blockIdx.x;
  const int b = blockIdx.y;
  const int tid = threadIdx.x;

  __shared__ float coef[BASIS];
  __shared__ float wbuf[CIN * TAPS];       // 4608 floats, indexed [i*9+tap]
  __shared__ float red[256];

  if (tid < BASIS) coef[tid] = shifts_coords[dirs[b] * BASIS + tid];
  __syncthreads();

  const float nrm    = sqrtf(normsq[b]);
  const float factor = shifts[b] / fmaxf(nrm, 1e-12f);
  const float scale  = rsqrtf((float)(CIN * KK * KK));   // 1/sqrt(4608)

  float ssq = 0.f;
  for (int e = tid; e < CIN * TAPS; e += 256) {
    int i = e / TAPS, tap = e % TAPS;
    long long p = ((long long)o * CIN + i) * TAPS + tap;
    float d = 0.f;
#pragma unroll
    for (int k = 0; k < BASIS; ++k) d += coef[k] * bv[(long long)k * PTOT + p];
    float w = scale * (weight[p] + d * factor) * s[b * CIN + i];
    wbuf[e] = w;
    ssq += w * w;
  }
  red[tid] = ssq;
  __syncthreads();
  for (int off = 128; off > 0; off >>= 1) {
    if (tid < off) red[tid] += red[tid + off];
    __syncthreads();
  }
  const float demod = rsqrtf(red[0] + 1e-8f);

  unsigned short* dst = wgt + ((long long)b * COUT + o) * (TAPS * CIN);
  for (int e2 = tid; e2 < CIN * TAPS; e2 += 256) {   // e2 = tap*512 + i  (contiguous write)
    int tap = e2 >> 9, i = e2 & 511;
    dst[e2] = f32_to_bf16(wbuf[i * TAPS + tap] * demod);
  }
}

// ---------------------------------------------------------------------------
// K2b: x -> padded, channel-last bf16 copy   xt[b][66][66][512]
// (halo rows/cols are pre-zeroed by hipMemsetAsync; this fills the interior)
// ---------------------------------------------------------------------------
__global__ void xpose_kernel(const float* __restrict__ x,
                             unsigned short* __restrict__ xt) {
  const int h = blockIdx.x;        // 0..63
  const int b = blockIdx.y;        // 0..15
  const int tid = threadIdx.x;

  __shared__ unsigned short t[64][68];   // [w][c-in-chunk], padded stride

  for (int cc = 0; cc < CIN; cc += 64) {
    __syncthreads();
    // read 64c x 64w  (float4 along w, coalesced)
    for (int e = tid; e < 64 * 16; e += 256) {
      int seg = e & 15, c = e >> 4;
      float4 v = *(const float4*)(x + (((long long)b * CIN + cc + c) * HH + h) * WW + seg * 4);
      t[seg * 4 + 0][c] = f32_to_bf16(v.x);
      t[seg * 4 + 1][c] = f32_to_bf16(v.y);
      t[seg * 4 + 2][c] = f32_to_bf16(v.z);
      t[seg * 4 + 3][c] = f32_to_bf16(v.w);
    }
    __syncthreads();
    // write: for each w, 64 consecutive channels (uint2 quads, coalesced)
    unsigned short* orow = xt + (((long long)b * HP + h + 1) * HP + 1) * CIN + cc;
    for (int e = tid; e < 64 * 16; e += 256) {
      int q = e & 15, w = e >> 4;
      uint2 vv = *(const uint2*)&t[w][q * 4];
      *(uint2*)(orow + (long long)w * CIN + q * 4) = vv;
    }
  }
}

// ---------------------------------------------------------------------------
// K3: grouped conv as 9 shifted GEMMs with v_wmma_f32_16x16x32_bf16.
// grid = (4 mtiles, 64 rows, 16 batches); 256 threads = 8 waves.
// Per-WAVE double-buffered A tiles via global_load_async_to_lds_b128:
// each wave stages only the 16 M-rows it consumes, so the 144-iteration
// tap loop needs no block barriers — just s_wait_asynccnt<=2.
// All four B fragments are loaded into distinct registers before the WMMAs
// so the scheduler can keep 8 ds_load_b128 in flight per tap.
// ---------------------------------------------------------------------------
#define XS_CSTR 40                     // padded channel stride (80B, 16B-aligned)
#define NTILES  (16 * TAPS)            // 144 A tiles per block

__global__ __launch_bounds__(256) void conv_kernel(const unsigned short* __restrict__ xt,
                                                   const unsigned short* __restrict__ wgt,
                                                   float* __restrict__ out) {
  const int mt = blockIdx.x;           // 0..3
  const int h  = blockIdx.y;           // 0..63
  const int b  = blockIdx.z;           // 0..15
  const int o_base = mt * 128;

  const int tid  = threadIdx.x;
  const int lane = tid & 31;
  const int wv   = tid >> 5;           // wave id 0..7
  const int half = (lane >> 4) & 1;
  const int ln16 = lane & 15;

  __shared__ __align__(16) unsigned short as_[2][128 * 32];         // A tiles [m][k]
  __shared__ __align__(16) unsigned short xs_[3 * HP * XS_CSTR];    // x slab [dh*66+widx][c]

  v8f acc[4];
#pragma unroll
  for (int nt = 0; nt < 4; ++nt) acc[nt] = (v8f){0.f,0.f,0.f,0.f,0.f,0.f,0.f,0.f};

  // per-WAVE A staging: wave wv owns rows [wv*16, wv*16+16).
  // lane covers (row r0 = lane>>2 and r0+8, 16B segment q0 = lane&3).
  const int r0 = lane >> 2;                // 0..7
  const int q0 = (lane & 3) * 8;           // ushort offset in 32-k row
  const unsigned short* wbaseA =
      wgt + ((long long)(b * COUT + o_base + wv * 16 + r0)) * (TAPS * CIN) + q0;
  const long long wstep8 = 8LL * (TAPS * CIN);        // +8 output rows

  unsigned short* adstA[2];
  adstA[0] = &as_[0][(wv * 16 + r0) * 32 + q0];
  adstA[1] = &as_[1][(wv * 16 + r0) * 32 + q0];

  // issue this wave's slice of A tile `lin` (chunk*9+tap) into buffer `buf`
  auto issue_A = [&](int lin2, int buf) {
    int tap = lin2 % TAPS;
    long long off = (long long)tap * CIN + (lin2 / TAPS) * 32;
    copy16_g2lds(wbaseA + off, adstA[buf]);
    copy16_g2lds(wbaseA + wstep8 + off, adstA[buf] + 8 * 32);
  };

  // x slab source base: padded row h+dh maps to image row h+dh-1
  const unsigned short* xrow = xt + ((long long)b * HP + h) * HP * CIN;

  issue_A(0, 0);                            // preload first tile (per wave)

  int lin = 0;
  for (int c0 = 0; c0 < CIN; c0 += 32) {
    __syncthreads();                        // all waves done reading prev xs
    // ---- async-stage x slab: 3*66 pixel-rows x 32 channels (4 b128 each) ----
    for (int e = tid; e < 3 * HP * 4; e += 256) {      // 792 lane-copies
      int q  = e & 3;
      int pr = e >> 2;                      // dh*66 + widx
      copy16_g2lds(xrow + (long long)pr * CIN + c0 + q * 8,
                   &xs_[pr * XS_CSTR + q * 8]);
    }
    WAIT_ASYNC(0);                          // xs + this chunk's tap-0 A landed
    __syncthreads();

    for (int tap = 0; tap < TAPS; ++tap, ++lin) {
      const int dh = tap / 3, dw = tap % 3;
      const int cur = lin & 1;

      if (lin + 1 < NTILES) {
        issue_A(lin + 1, cur ^ 1);          // prefetch next tile (this wave)
        WAIT_ASYNC(2);                      // in-order: tile `lin` has landed
      } else {
        WAIT_ASYNC(0);                      // last tile: drain
      }

      // ---- A fragment (contiguous k = 16*half + t, same order as B) ----
      const unsigned short* arow = &as_[cur][(wv * 16 + ln16) * 32];
      FragA af;
      af.u[0] = *(const us8*)(arow + 16 * half);         // t=0..7  : k=16*half+t
      af.u[1] = *(const us8*)(arow + 16 * half + 8);     // t=8..15 : k=16*half+t

      // ---- load all 4 B fragments first (distinct regs -> pipelined ds) ----
      FragA bfr[4];
#pragma unroll
      for (int nt = 0; nt < 4; ++nt) {
        int widx = nt * 16 + ln16 + dw;                  // padded w index
        const unsigned short* bp = &xs_[(dh * HP + widx) * XS_CSTR + 16 * half];
        bfr[nt].u[0] = *(const us8*)(bp);                // k = 16*half + 0..7
        bfr[nt].u[1] = *(const us8*)(bp + 8);            // k = 16*half + 8..15
      }
      // ---- 4 WMMAs sharing the A fragment ----
#pragma unroll
      for (int nt = 0; nt < 4; ++nt) {
        acc[nt] = __builtin_amdgcn_wmma_f32_16x16x32_bf16(
            /*neg_a=*/false, af.bf, /*neg_b=*/false, bfr[nt].bf,
            /*c_mod=*/(short)0, acc[nt], /*reuse_a=*/false, /*reuse_b=*/false);
      }
    }
  }

  // ---- epilogue: C/D layout -> out[b][o][h][w] ----
#pragma unroll
  for (int nt = 0; nt < 4; ++nt) {
    int w = nt * 16 + ln16;
#pragma unroll
    for (int v = 0; v < 8; ++v) {
      int o = o_base + wv * 16 + v + 8 * half;    // VGPR v: M=v (lanes 0-15), M=v+8 (16-31)
      out[(((long long)b * COUT + o) * HH + h) * WW + w] = acc[nt][v];
    }
  }
}

// ---------------------------------------------------------------------------
extern "C" void kernel_launch(void* const* d_in, const int* in_sizes, int n_in,
                              void* d_out, int out_size, void* d_ws, size_t ws_size,
                              hipStream_t stream) {
  (void)in_sizes; (void)n_in; (void)out_size; (void)ws_size;

  const float* x             = (const float*)d_in[0];
  const float* style         = (const float*)d_in[1];
  const float* weight        = (const float*)d_in[2];
  const float* mod_w         = (const float*)d_in[3];
  const float* mod_b         = (const float*)d_in[4];
  const float* bv            = (const float*)d_in[5];
  const float* shifts_coords = (const float*)d_in[6];
  const float* batch_shifts  = (const float*)d_in[7];
  const int*   batch_dirs    = (const int*)d_in[8];
  float* out = (float*)d_out;

  // workspace layout (bytes)
  char* wsb = (char*)d_ws;
  float* s_ws   = (float*)wsb;                              // 32768 B
  float* normsq = (float*)(wsb + 32768);                    // 64 B (16 used)
  unsigned short* wgt = (unsigned short*)(wsb + 33024);     // 75,497,472 B
  unsigned short* xt  = (unsigned short*)(wsb + 33024 + 75497472);  // 71,368,704 B

  (void)hipMemsetAsync(normsq, 0, BB * sizeof(float), stream);
  (void)hipMemsetAsync(xt, 0, (long long)BB * HP * HP * CIN * sizeof(unsigned short), stream);

  style_kernel<<<(BB * CIN) / 256, 256, 0, stream>>>(style, mod_w, mod_b, s_ws);

  normsq_kernel<<<1024, 256, 0, stream>>>(bv, shifts_coords, batch_dirs, normsq);

  xpose_kernel<<<dim3(HH, BB), 256, 0, stream>>>(x, xt);

  build_weights_kernel<<<dim3(COUT, BB), 256, 0, stream>>>(
      weight, bv, shifts_coords, batch_dirs, batch_shifts, s_ws, normsq, wgt);

  conv_kernel<<<dim3(COUT / 128, HH, BB), 256, 0, stream>>>(xt, wgt, out);
}
